// FixedSpectralBlock_16716012716515
// MI455X (gfx1250) — matrix operand order repaired and checked by reading, not compiled, not run on hardware
//
#include <hip/hip_runtime.h>
#include <hip/hip_bf16.h>
#include <math.h>

typedef __attribute__((ext_vector_type(16))) _Float16 v16h;
typedef __attribute__((ext_vector_type(8)))  _Float16 v8h;
typedef __attribute__((ext_vector_type(8)))  float    v8f;

#define NTHR 256

// ---------------------------------------------------------------------------
// gfx1250 async global->LDS copy (16B per lane), tracked by ASYNCcnt.
// VDST = per-lane LDS byte offset, VADDR = per-lane 64-bit global address.
// ---------------------------------------------------------------------------
__device__ __forceinline__ void async_g2l_b128(const _Float16* g, uint32_t ldsOff)
{
    uint64_t ga = (uint64_t)(uintptr_t)g;
    asm volatile("global_load_async_to_lds_b128 %0, %1, off"
                 :: "v"(ldsOff), "v"(ga) : "memory");
}
__device__ __forceinline__ void wait_async0()
{
#if __has_builtin(__builtin_amdgcn_s_wait_asynccnt)
    __builtin_amdgcn_s_wait_asynccnt(0);
#else
    asm volatile("s_wait_asynccnt 0x0" ::: "memory");
#endif
}

// ---------------------------------------------------------------------------
// Kernel 1: per-row LayerNorm stats (mean, rstd) over C.  rows = B*T
// ---------------------------------------------------------------------------
__global__ __launch_bounds__(NTHR) void ln_stats_kernel(
    const float* __restrict__ x, float* __restrict__ mu, float* __restrict__ rs, int C)
{
    int row = blockIdx.x;
    const float* xr = x + (size_t)row * C;
    int tid = threadIdx.x;
    float s = 0.f, s2 = 0.f;
    for (int c = tid; c < C; c += NTHR) { float v = xr[c]; s += v; s2 += v * v; }
    __shared__ float r1[NTHR], r2[NTHR];
    r1[tid] = s; r2[tid] = s2; __syncthreads();
    for (int off = NTHR / 2; off > 0; off >>= 1) {
        if (tid < off) { r1[tid] += r1[tid + off]; r2[tid] += r2[tid + off]; }
        __syncthreads();
    }
    if (tid == 0) {
        float m = r1[0] / C;
        float var = r2[0] / C - m * m;
        mu[row] = m;
        rs[row] = rsqrtf(var + 1e-5f);
    }
}

// ---------------------------------------------------------------------------
// Kernel 2: apply LN and transpose [b][t][c] -> [b][c][t] (32x64 LDS tiles)
// ---------------------------------------------------------------------------
__global__ __launch_bounds__(NTHR) void transpose_norm_kernel(
    const float* __restrict__ x, const float* __restrict__ mu, const float* __restrict__ rs,
    const float* __restrict__ w, const float* __restrict__ bias,
    float* __restrict__ xnT, int T, int C)
{
    __shared__ float tile[32][65];
    int t0 = blockIdx.x * 32, c0 = blockIdx.y * 64, b = blockIdx.z;
    const float* xb = x + (size_t)b * T * C;
    for (int i = threadIdx.x; i < 2048; i += NTHR) {
        int r = i >> 6, cc = i & 63;
        int t = t0 + r, c = c0 + cc;
        float v = xb[(size_t)t * C + c];
        int gr = b * T + t;
        tile[r][cc] = (v - mu[gr]) * rs[gr] * w[c] + bias[c];
    }
    __syncthreads();
    float* ob = xnT + (size_t)b * C * T;
    for (int i = threadIdx.x; i < 2048; i += NTHR) {
        int crow = i >> 5, tcol = i & 31;
        ob[(size_t)(c0 + crow) * T + t0 + tcol] = tile[tcol][crow];
    }
}

// ---------------------------------------------------------------------------
// Kernel 3: pooled[b*C+c] = mean over t of xnT[b][c][t]
// ---------------------------------------------------------------------------
__global__ __launch_bounds__(NTHR) void pool_kernel(
    const float* __restrict__ xnT, float* __restrict__ pooled, int T)
{
    int bc = blockIdx.x;
    const float* rp = xnT + (size_t)bc * T;
    int tid = threadIdx.x;
    float s = 0.f;
    for (int t = tid; t < T; t += NTHR) s += rp[t];
    __shared__ float r1[NTHR];
    r1[tid] = s; __syncthreads();
    for (int off = NTHR / 2; off > 0; off >>= 1) {
        if (tid < off) r1[tid] += r1[tid + off];
        __syncthreads();
    }
    if (tid == 0) pooled[bc] = r1[0] / T;
}

// ---------------------------------------------------------------------------
// Kernel 4: context gate  g = sigmoid(pooled @ Wctx + bctx)
// ---------------------------------------------------------------------------
__global__ __launch_bounds__(NTHR) void gctx_kernel(
    const float* __restrict__ pooled, const float* __restrict__ W,
    const float* __restrict__ bb, float* __restrict__ g, int C)
{
    int idx = blockIdx.x * NTHR + threadIdx.x;
    int b = idx / C, c = idx - b * C;
    const float* p = pooled + (size_t)b * C;
    float s = bb[c];
    for (int k = 0; k < C; k++) s += p[k] * W[(size_t)k * C + c];
    g[idx] = 1.f / (1.f + expf(-s));
}

// ---------------------------------------------------------------------------
// Kernel 5: filter spectrum H[f]=rfft(kernel,N)[f]*sigmoid(logits[f])*mask[f],
// stored Hermitian-extended over N bins in BIT-REVERSED order.
// ---------------------------------------------------------------------------
__global__ __launch_bounds__(NTHR) void kfreq_kernel(
    const float* __restrict__ kern, const float* __restrict__ logits,
    const int* __restrict__ cutp, float2* __restrict__ Hbr,
    int K, int Fbins, int N, int LOGN)
{
    int f = blockIdx.x * NTHR + threadIdx.x;
    if (f >= Fbins) return;
    float re = 0.f, im = 0.f;
    float inv = 2.0f / (float)N;
    for (int k = 0; k < K; k++) {
        int m = (f * k) & (N - 1);
        float s, c;
        sincospif(-(float)m * inv, &s, &c);
        float kv = kern[k];
        re += kv * c; im += kv * s;
    }
    float gf = 1.f / (1.f + expf(-logits[f]));
    int cut = *cutp;
    int cc = cut < Fbins ? cut : Fbins;
    float mask = 1.f;
    if (cc < Fbins) {
        int tr = cc < 16 ? cc : 16;
        if (f >= cc) mask = 0.f;
        else if (f >= cc - tr) {
            int i = f - (cc - tr);
            float ti = (tr > 1) ? (float)i / (float)(tr - 1) : 0.f;
            mask = 0.5f * (1.f + cospif(ti));
        }
    }
    float sc = gf * mask;
    re *= sc; im *= sc;
    int rev = (int)(__brev((unsigned)f) >> (32 - LOGN));
    Hbr[rev] = make_float2(re, im);
    if (f > 0 && f < N / 2) {
        int rev2 = (int)(__brev((unsigned)(N - f)) >> (32 - LOGN));
        Hbr[rev2] = make_float2(re, -im);
    }
}

// ---------------------------------------------------------------------------
// Kernel 6: channel-pair packed FFT convolution.
// z = xn[c0] + i*xn[c1]; DIF fwd (bit-rev out) -> *Hbr (bit-rev) -> DIT inv.
// x2 = x + gain*g_ctx * y.     LDS: re/im (64KB) + twiddles (32KB) = 96KB.
// ---------------------------------------------------------------------------
__global__ __launch_bounds__(NTHR) void fftconv_kernel(
    const float* __restrict__ xnT, const float2* __restrict__ Hbr,
    const float* __restrict__ x, const float* __restrict__ gain,
    const float* __restrict__ gctx, float* __restrict__ x2,
    int T, int C)
{
    const int N = 8192, HN = 4096;
    extern __shared__ float sm[];
    float* re  = sm;
    float* im  = sm + N;
    float* twr = sm + 2 * N;
    float* twi = sm + 2 * N + HN;
    int tid = threadIdx.x;
    int bc = blockIdx.x;
    int b = bc / (C / 2);
    int pr = bc - b * (C / 2);
    int c0 = pr * 2, c1 = c0 + 1;

    const float* r0 = xnT + ((size_t)b * C + c0) * T;
    const float* r1 = xnT + ((size_t)b * C + c1) * T;
    for (int t = tid; t < T; t += NTHR) { re[t] = r0[t]; im[t] = r1[t]; }
    for (int t = T + tid; t < N; t += NTHR) { re[t] = 0.f; im[t] = 0.f; }
    for (int j = tid; j < HN; j += NTHR) {
        float s, c;
        sincospif(-(float)j * (2.0f / (float)N), &s, &c);
        twr[j] = c; twi[j] = s;
    }
    __syncthreads();

    // forward radix-2 DIF (natural in, bit-reversed out)
    for (int len = N; len >= 2; len >>= 1) {
        int h = len >> 1;
        int tstep = N / len;
        for (int q = tid; q < HN; q += NTHR) {
            int j = q & (h - 1);
            int i0 = 2 * q - j;
            int i1 = i0 + h;
            float ur = re[i0], ui = im[i0], vr = re[i1], vi = im[i1];
            int tw = j * tstep;
            float wr = twr[tw], wi = twi[tw];
            float br = ur - vr, bi = ui - vi;
            re[i0] = ur + vr;            im[i0] = ui + vi;
            re[i1] = br * wr - bi * wi;  im[i1] = br * wi + bi * wr;
        }
        __syncthreads();
    }

    // pointwise multiply with bit-reversed Hermitian filter
    for (int n = tid; n < N; n += NTHR) {
        float2 hv = Hbr[n];
        float zr = re[n], zi = im[n];
        re[n] = zr * hv.x - zi * hv.y;
        im[n] = zr * hv.y + zi * hv.x;
    }
    __syncthreads();

    // inverse radix-2 DIT (bit-reversed in, natural out), conj twiddles
    for (int len = 2; len <= N; len <<= 1) {
        int h = len >> 1;
        int tstep = N / len;
        for (int q = tid; q < HN; q += NTHR) {
            int j = q & (h - 1);
            int i0 = 2 * q - j;
            int i1 = i0 + h;
            int tw = j * tstep;
            float wr = twr[tw], wi = -twi[tw];
            float vr = re[i1], vi = im[i1];
            float tr_ = vr * wr - vi * wi;
            float ti_ = vr * wi + vi * wr;
            float ur = re[i0], ui = im[i0];
            re[i0] = ur + tr_; im[i0] = ui + ti_;
            re[i1] = ur - tr_; im[i1] = ui - ti_;
        }
        __syncthreads();
    }

    float s0 = gain[c0] * gctx[b * C + c0] * (1.0f / (float)N);
    float s1 = gain[c1] * gctx[b * C + c1] * (1.0f / (float)N);
    const float* xb = x  + (size_t)b * T * C;
    float*       ob = x2 + (size_t)b * T * C;
    for (int t = tid; t < T; t += NTHR) {
        size_t o = (size_t)t * C + c0;
        ob[o]     = xb[o]     + re[t] * s0;
        ob[o + 1] = xb[o + 1] + im[t] * s1;
    }
}

// ---------------------------------------------------------------------------
// Kernel 7: FFN LayerNorm over C, output f16 activation matrix [BT, C]
// ---------------------------------------------------------------------------
__global__ __launch_bounds__(NTHR) void ffn_ln_kernel(
    const float* __restrict__ x2, const float* __restrict__ w,
    const float* __restrict__ bias, _Float16* __restrict__ Ah, int C)
{
    int row = blockIdx.x;
    const float* xr = x2 + (size_t)row * C;
    int tid = threadIdx.x;
    float s = 0.f, s2 = 0.f;
    for (int c = tid; c < C; c += NTHR) { float v = xr[c]; s += v; s2 += v * v; }
    __shared__ float r1[NTHR], r2[NTHR];
    __shared__ float mm, ri;
    r1[tid] = s; r2[tid] = s2; __syncthreads();
    for (int off = NTHR / 2; off > 0; off >>= 1) {
        if (tid < off) { r1[tid] += r1[tid + off]; r2[tid] += r2[tid + off]; }
        __syncthreads();
    }
    if (tid == 0) {
        float m = r1[0] / C;
        float var = r2[0] / C - m * m;
        mm = m; ri = rsqrtf(var + 1e-5f);
    }
    __syncthreads();
    float m = mm, rr = ri;
    for (int c = tid; c < C; c += NTHR)
        Ah[(size_t)row * C + c] = (_Float16)((xr[c] - m) * rr * w[c] + bias[c]);
}

// ---------------------------------------------------------------------------
// Kernel 8: f32 -> f16 conversion (weights)
// ---------------------------------------------------------------------------
__global__ __launch_bounds__(NTHR) void cvt_half_kernel(
    const float* __restrict__ src, _Float16* __restrict__ dst, int n)
{
    int i = blockIdx.x * NTHR + threadIdx.x;
    if (i < n) dst[i] = (_Float16)src[i];
}

// ---------------------------------------------------------------------------
// WMMA GEMM core: 128x128 block, BK=32, 8 waves x (2x4) 16x16 WMMA tiles.
// Double-buffered LDS; A tile via async global->LDS (ASYNCcnt), B tile via
// VGPR + transposed scatter (WMMA B operand wants K-contiguous per lane).
// ---------------------------------------------------------------------------
__device__ __forceinline__ void gemm_tile_mma(
    const _Float16* __restrict__ A, const _Float16* __restrict__ Bm,
    _Float16 (&sA)[2][4096], _Float16 (&sB)[2][4096], v8f acc[2][4],
    int blockM, int blockN, int K, int N)
{
    int tid  = threadIdx.x;
    int lane = tid & 31;
    int wave = tid >> 5;
    int wm = wave >> 1, wn = wave & 1;
    int hsel = lane >> 4, ml = lane & 15;
    int am = tid >> 1,  ak = (tid & 1) * 16;
    int bk = tid >> 3,  bn = (tid & 7) * 16;

    const _Float16* gA0 = A  + (size_t)(blockM + am) * K + ak;
    const _Float16* gB0 = Bm + (size_t)bk * N + blockN + bn;
    uint32_t ldsA0 = (uint32_t)(uintptr_t)&sA[0][am * 32 + ak];
    uint32_t ldsA1 = (uint32_t)(uintptr_t)&sA[1][am * 32 + ak];

    // prologue: stage tile 0
    async_g2l_b128(gA0,     ldsA0);
    async_g2l_b128(gA0 + 8, ldsA0 + 16);
    {
        v8h b0 = *(const v8h*)gB0;
        v8h b1 = *(const v8h*)(gB0 + 8);
#pragma unroll
        for (int i = 0; i < 8; i++) {
            sB[0][(bn + i) * 32 + bk]     = b0[i];
            sB[0][(bn + 8 + i) * 32 + bk] = b1[i];
        }
    }
    wait_async0();
    __syncthreads();

    int cur = 0;
    for (int kt = 0; kt < K; kt += 32, cur ^= 1) {
        int alt = cur ^ 1;
        bool hasNext = (kt + 32) < K;
        v8h nb0 = {}, nb1 = {};
        if (hasNext) {
            const _Float16* gA = gA0 + kt + 32;
            uint32_t dst = alt ? ldsA1 : ldsA0;
            async_g2l_b128(gA,     dst);
            async_g2l_b128(gA + 8, dst + 16);
            const _Float16* gB = gB0 + (size_t)(kt + 32) * N;
            nb0 = *(const v8h*)gB;
            nb1 = *(const v8h*)(gB + 8);
            __builtin_prefetch(gB + (size_t)32 * N, 0, 3);   // tile kt+64, near cache
        }

        // fragments from current buffers
        v16h af[2], bf[4];
#pragma unroll
        for (int mi = 0; mi < 2; mi++) {
            const _Float16* p = &sA[cur][(wm * 32 + mi * 16 + ml) * 32];
            v8h lo = *(const v8h*)(p + 8 * hsel);
            v8h hi = *(const v8h*)(p + 16 + 8 * hsel);
            af[mi] = __builtin_shufflevector(lo, hi, 0, 1, 2, 3, 4, 5, 6, 7,
                                             8, 9, 10, 11, 12, 13, 14, 15);
        }
#pragma unroll
        for (int ni = 0; ni < 4; ni++) {
            const _Float16* p = &sB[cur][(wn * 64 + ni * 16 + ml) * 32];
            v8h lo = *(const v8h*)(p + 8 * hsel);
            v8h hi = *(const v8h*)(p + 16 + 8 * hsel);
            bf[ni] = __builtin_shufflevector(lo, hi, 0, 1, 2, 3, 4, 5, 6, 7,
                                             8, 9, 10, 11, 12, 13, 14, 15);
        }
#pragma unroll
        for (int mi = 0; mi < 2; mi++)
#pragma unroll
            for (int ni = 0; ni < 4; ni++)
                acc[mi][ni] = __builtin_amdgcn_wmma_f32_16x16x32_f16(
                    false, af[mi], false, bf[ni], (short)0, acc[mi][ni], false, false);

        if (hasNext) {
#pragma unroll
            for (int i = 0; i < 8; i++) {
                sB[alt][(bn + i) * 32 + bk]     = nb0[i];
                sB[alt][(bn + 8 + i) * 32 + bk] = nb1[i];
            }
        }
        wait_async0();       // A tile for next iter landed in LDS
        __syncthreads();     // publish alt buffers / retire cur reads
    }
}

// GEMM1: Hout = gelu(A @ Bm + bias)  -> f16
__global__ __launch_bounds__(NTHR) void gemm1_gelu_kernel(
    const _Float16* __restrict__ A, const _Float16* __restrict__ Bm,
    const float* __restrict__ bias, _Float16* __restrict__ Hout,
    int K, int N)
{
    __shared__ _Float16 sA[2][4096];
    __shared__ _Float16 sB[2][4096];
    int blockN = blockIdx.x * 128;
    int blockM = blockIdx.y * 128;
    v8f acc[2][4] = {};
    gemm_tile_mma(A, Bm, sA, sB, acc, blockM, blockN, K, N);

    int lane = threadIdx.x & 31, wave = threadIdx.x >> 5;
    int wm = wave >> 1, wn = wave & 1;
    int hsel = lane >> 4, ml = lane & 15;
#pragma unroll
    for (int mi = 0; mi < 2; mi++)
#pragma unroll
        for (int ni = 0; ni < 4; ni++) {
            int col = blockN + wn * 64 + ni * 16 + ml;
            float bv = bias[col];
#pragma unroll
            for (int r = 0; r < 8; r++) {
                int row = blockM + wm * 32 + mi * 16 + r + 8 * hsel;
                float v = acc[mi][ni][r] + bv;
                float g = 0.5f * v * (1.0f + erff(v * 0.70710678118f));
                Hout[(size_t)row * N + col] = (_Float16)g;
            }
        }
}

// GEMM2: out = resid + A @ Bm + bias  -> f32
__global__ __launch_bounds__(NTHR) void gemm2_res_kernel(
    const _Float16* __restrict__ A, const _Float16* __restrict__ Bm,
    const float* __restrict__ bias, const float* __restrict__ resid,
    float* __restrict__ out, int K, int N)
{
    __shared__ _Float16 sA[2][4096];
    __shared__ _Float16 sB[2][4096];
    int blockN = blockIdx.x * 128;
    int blockM = blockIdx.y * 128;
    v8f acc[2][4] = {};
    gemm_tile_mma(A, Bm, sA, sB, acc, blockM, blockN, K, N);

    int lane = threadIdx.x & 31, wave = threadIdx.x >> 5;
    int wm = wave >> 1, wn = wave & 1;
    int hsel = lane >> 4, ml = lane & 15;
#pragma unroll
    for (int mi = 0; mi < 2; mi++)
#pragma unroll
        for (int ni = 0; ni < 4; ni++) {
            int col = blockN + wn * 64 + ni * 16 + ml;
            float bv = bias[col];
#pragma unroll
            for (int r = 0; r < 8; r++) {
                int row = blockM + wm * 32 + mi * 16 + r + 8 * hsel;
                size_t o = (size_t)row * N + col;
                out[o] = resid[o] + acc[mi][ni][r] + bv;
            }
        }
}

// ---------------------------------------------------------------------------
extern "C" void kernel_launch(void* const* d_in, const int* in_sizes, int n_in,
                              void* d_out, int out_size, void* d_ws, size_t ws_size,
                              hipStream_t stream) {
    const float* x       = (const float*)d_in[0];
    const float* kern    = (const float*)d_in[1];
    const float* gain    = (const float*)d_in[2];
    const float* logits  = (const float*)d_in[3];
    const float* gctx_w  = (const float*)d_in[4];
    const float* gctx_b  = (const float*)d_in[5];
    const float* ln_w    = (const float*)d_in[6];
    const float* ln_b    = (const float*)d_in[7];
    const float* fln_w   = (const float*)d_in[8];
    const float* fln_b   = (const float*)d_in[9];
    const float* w1      = (const float*)d_in[10];
    const float* b1      = (const float*)d_in[11];
    const float* w2      = (const float*)d_in[12];
    const float* b2      = (const float*)d_in[13];
    const int*   cutoff  = (const int*)d_in[14];
    float* out = (float*)d_out;

    const int C     = in_sizes[2];           // 1024
    const int K     = in_sizes[1];           // 1024
    const int Fbins = in_sizes[3];           // 4097
    const int Nfft  = (Fbins - 1) * 2;       // 8192
    int logn = 0; while ((1 << logn) < Nfft) logn++;
    const int Hd    = in_sizes[11];          // 2048
    const int BT    = in_sizes[0] / C;       // 16384
    const int T     = 4096;
    const int B     = BT / T;

    // workspace layout
    char* w = (char*)d_ws;
    const size_t MB = 1u << 20;
    float*    xnT   = (float*)(w);                   // B*C*T f32     = 64MB
    float*    x2    = (float*)(w + 64 * MB);         // B*T*C f32     = 64MB
    _Float16* Ah    = (_Float16*)(w + 128 * MB);     // BT*C f16      = 32MB
    _Float16* W1h   = (_Float16*)(w + 160 * MB);     // C*Hd f16      = 4MB
    _Float16* W2h   = (_Float16*)(w + 164 * MB);     // Hd*C f16      = 4MB
    float2*   Hbr   = (float2*)(w + 168 * MB);       // Nfft cpx      = 64KB
    float*    mu    = (float*)(w + 168 * MB + (64u << 10));
    float*    rsb   = mu + BT;
    float*    pooled= rsb + BT;
    float*    gct   = pooled + B * C;
    _Float16* Hg    = (_Float16*)xnT;                // reuse: BT*Hd f16 = 64MB
    size_t need = 168 * MB + (64u << 10) + (size_t)(2 * BT + 2 * B * C) * 4 + 256;
    if (ws_size < need) return;

    // 1) LN stats
    ln_stats_kernel<<<BT, NTHR, 0, stream>>>(x, mu, rsb, C);
    // 2) normalize + transpose to [b][c][t]
    transpose_norm_kernel<<<dim3(T / 32, C / 64, B), NTHR, 0, stream>>>(
        x, mu, rsb, ln_w, ln_b, xnT, T, C);
    // 3) pooled mean over t
    pool_kernel<<<B * C, NTHR, 0, stream>>>(xnT, pooled, T);
    // 4) context gate
    gctx_kernel<<<(B * C) / NTHR, NTHR, 0, stream>>>(pooled, gctx_w, gctx_b, gct, C);
    // 5) filter spectrum (bit-reversed, Hermitian-extended)
    kfreq_kernel<<<(Fbins + NTHR - 1) / NTHR, NTHR, 0, stream>>>(
        kern, logits, cutoff, Hbr, K, Fbins, Nfft, logn);
    // 6) packed-pair FFT convolution + residual -> x2
    fftconv_kernel<<<B * C / 2, NTHR, 96 * 1024, stream>>>(
        xnT, Hbr, x, gain, gct, x2, T, C);
    // 7) FFN LN -> f16 activations
    ffn_ln_kernel<<<BT, NTHR, 0, stream>>>(x2, fln_w, fln_b, Ah, C);
    // 8) weight conversion
    cvt_half_kernel<<<(C * Hd + NTHR - 1) / NTHR, NTHR, 0, stream>>>(w1, W1h, C * Hd);
    cvt_half_kernel<<<(Hd * C + NTHR - 1) / NTHR, NTHR, 0, stream>>>(w2, W2h, Hd * C);
    // 9) GEMM1 + GELU  (WMMA, async A path)
    gemm1_gelu_kernel<<<dim3(Hd / 128, BT / 128), NTHR, 0, stream>>>(Ah, W1h, b1, Hg, C, Hd);
    // 10) GEMM2 + bias + residual (WMMA, async A path)
    gemm2_res_kernel<<<dim3(C / 128, BT / 128), NTHR, 0, stream>>>(Hg, W2h, b2, x2, out, Hd, C);
}
